// MultiHeadAttention_12051678232860
// MI455X (gfx1250) — matrix operand order repaired and checked by reading, not compiled, not run on hardware
//
#include <hip/hip_runtime.h>
#include <hip/hip_bf16.h>
#include <cstdint>
#include <cstddef>

#define B_    4
#define NH_   8
#define S_    2048
#define HD_   64
#define H_    512
#define MTOT_ (B_ * S_)          // 8192 rows for the projection GEMMs

typedef __attribute__((ext_vector_type(16))) _Float16 v16h;
typedef __attribute__((ext_vector_type(8)))  _Float16 v8h;
typedef __attribute__((ext_vector_type(4)))  _Float16 v4h;
typedef __attribute__((ext_vector_type(8)))  float    v8f;
typedef __attribute__((ext_vector_type(4)))  float    v4f;
typedef __attribute__((ext_vector_type(4)))  unsigned int u32x4;
typedef __attribute__((ext_vector_type(8)))  int      i32x8;
typedef __attribute__((ext_vector_type(4)))  int      i32x4;

#define ZERO8 {0.f,0.f,0.f,0.f,0.f,0.f,0.f,0.f}

static __device__ __forceinline__ v8f wmma_f16(v16h a, v16h b, v8f c) {
  return __builtin_amdgcn_wmma_f32_16x16x32_f16(false, a, false, b, (short)0, c,
                                                false, false);
}

static __device__ __forceinline__ v16h cat8(v8h lo, v8h hi) {
  v16h r;
#pragma unroll
  for (int i = 0; i < 8; ++i) { r[i] = lo[i]; r[8 + i] = hi[i]; }
  return r;
}

static __device__ __forceinline__ float redmax16(float v) {
#pragma unroll
  for (int m = 1; m < 16; m <<= 1) v = fmaxf(v, __shfl_xor(v, m, 32));
  return v;
}
static __device__ __forceinline__ float redsum16(float v) {
#pragma unroll
  for (int m = 1; m < 16; m <<= 1) v += __shfl_xor(v, m, 32);
  return v;
}

// ---------------------------------------------------------------------------
// TDM: load a 2D f16 tile (tile_w x tile_h elements) from a row-major tensor
// (tensor_w x tensor_h, row stride row_stride elements) into LDS at lds_off,
// inserting pad_amount per pad_interval (codes per D# Group1) to produce a
// padded, bank-conflict-free LDS row stride. D# packed per CDNA5 ISA §8.3/8.4.
// Issue from ONE wave only; completion tracked on TENSORcnt.
// This toolchain exposes the 6-arg builtin: (g0 v4u, g1 v8i, v4i, v4i, v8i, cpol).
// ---------------------------------------------------------------------------
static __device__ __forceinline__ void tdm_load_2d_f16(
    const _Float16* gsrc, unsigned int lds_off,
    unsigned int tile_w, unsigned int tile_h,
    unsigned int tensor_w, unsigned int tensor_h, unsigned int row_stride,
    unsigned int pad_interval_code, unsigned int pad_amount_code) {
  unsigned long long ga = (unsigned long long)(uintptr_t)gsrc;
  u32x4 g0;
  g0[0] = 1u;                                   // count=1, user mode, no gather
  g0[1] = lds_off;                              // lds_addr
  g0[2] = (unsigned int)ga;                     // global_addr[31:0]
  g0[3] = (unsigned int)((ga >> 32) & 0x1FFFFFFu) | (2u << 30);  // addr[56:32], type=2
  i32x8 g1;
  g1[0] = (int)((1u << 16) |                    // data_size = 2 bytes
                (1u << 20) |                    // pad_enable
                (pad_interval_code << 22) | (pad_amount_code << 25));
  g1[1] = (int)(tensor_w << 16);                // tensor_dim0[15:0]
  g1[2] = (int)((tensor_w >> 16) | (tensor_h << 16));   // dim0 hi | dim1 lo
  g1[3] = (int)((tensor_h >> 16) | (tile_w << 16));     // dim1 hi | tile_dim0
  g1[4] = (int)tile_h;                          // tile_dim1 (tile_dim2 = 0)
  g1[5] = (int)row_stride;                      // tensor_dim0_stride[31:0]
  g1[6] = 0;
  g1[7] = 0;
  i32x4 z4 = {0, 0, 0, 0};                      // groups 2/3: zeros (2-D tile)
  i32x8 z8 = {0, 0, 0, 0, 0, 0, 0, 0};          // trailing group: zero-filled
  __builtin_amdgcn_tensor_load_to_lds(g0, g1, z4, z4, z8, 0);
}

static __device__ __forceinline__ unsigned int lds_off_of(const void* p) {
  return (unsigned int)(uintptr_t)p;            // generic LDS VA: low 32 bits = LDS offset
}

// ---------------------------------------------------------------------------
// Kernel 1: X(8192x512 f32) @ W^T(512x512 f32) + b  ->  f16, head-split layout.
// (f32->f16 conversion fused in staging; TDM moves raw bytes, so staging stays VALU.)
// ---------------------------------------------------------------------------
__global__ __launch_bounds__(256) void qkv_proj_kernel(
    const float* __restrict__ X, const float* __restrict__ W,
    const float* __restrict__ bias, _Float16* __restrict__ Out, int transpose_v) {
  __shared__ __align__(16) _Float16 Xs[128 * 40];
  __shared__ __align__(16) _Float16 Ws[64 * 40];

  const int t = threadIdx.x, lane = t & 31, wave = t >> 5;
  const int l16 = lane & 15, half = lane >> 4;
  const int Mbase = blockIdx.x * 128, Nbase = blockIdx.y * 64;

  v8f acc[4] = {ZERO8, ZERO8, ZERO8, ZERO8};

  for (int k0 = 0; k0 < H_; k0 += 32) {
#pragma unroll
    for (int rr = 0; rr < 4; ++rr) {
      int row = rr * 32 + (t >> 3);
      int c4 = (t & 7) * 4;
      v4f x = *(const v4f*)&X[(size_t)(Mbase + row) * H_ + k0 + c4];
      v4h hx;
#pragma unroll
      for (int i = 0; i < 4; ++i) hx[i] = (_Float16)x[i];
      *(v4h*)&Xs[row * 40 + c4] = hx;
    }
#pragma unroll
    for (int rr = 0; rr < 2; ++rr) {
      int row = rr * 32 + (t >> 3);
      int c4 = (t & 7) * 4;
      v4f x = *(const v4f*)&W[(size_t)(Nbase + row) * H_ + k0 + c4];
      v4h hx;
#pragma unroll
      for (int i = 0; i < 4; ++i) hx[i] = (_Float16)x[i];
      *(v4h*)&Ws[row * 40 + c4] = hx;
    }
    __syncthreads();

    int ar = wave * 16 + l16;
    int akl = half * 8;
    v16h a = cat8(*(const v8h*)&Xs[ar * 40 + akl],
                  *(const v8h*)&Xs[ar * 40 + akl + 16]);
#pragma unroll
    for (int nt = 0; nt < 4; ++nt) {
      int br = nt * 16 + l16;
      int bkl = half * 8;
      v16h bfr = cat8(*(const v8h*)&Ws[br * 40 + bkl],
                      *(const v8h*)&Ws[br * 40 + bkl + 16]);
      acc[nt] = wmma_f16(a, bfr, acc[nt]);
    }
    __syncthreads();
  }

  if (!transpose_v) {
#pragma unroll
    for (int nt = 0; nt < 4; ++nt) {
      int n = Nbase + nt * 16 + l16;
      int h = n >> 6, d = n & 63;
      float bv = bias[n];
#pragma unroll
      for (int i = 0; i < 8; ++i) {
        int M = Mbase + wave * 16 + half * 8 + i;
        int b = M >> 11, s = M & (S_ - 1);
        Out[(((size_t)(b * NH_ + h) * S_ + s) << 6) + d] =
            (_Float16)(acc[nt][i] + bv);
      }
    }
  } else {
#pragma unroll
    for (int nt = 0; nt < 4; ++nt) {
      int n = Nbase + nt * 16 + l16;
      int h = n >> 6, d = n & 63;
      float bv = bias[n];
      int M0 = Mbase + wave * 16 + half * 8;
      int b = M0 >> 11, s0 = M0 & (S_ - 1);
      v8h pk;
#pragma unroll
      for (int i = 0; i < 8; ++i) pk[i] = (_Float16)(acc[nt][i] + bv);
      *(v8h*)&Out[(((size_t)(b * NH_ + h) * HD_ + d) << 11) + s0] = pk;
    }
  }
}

// ---------------------------------------------------------------------------
// Kernel 2: fused attention with TDM-staged K/V tiles.
//  - K tiles 32x64 -> LDS stride 72 halves (144B: conflict-free), triple-buffered
//  - V tiles 64x32 (from V^T) -> LDS stride 40 halves, triple-buffered
//  - wave 0 issues tensor_load_to_lds with prefetch distance 2; barrier publishes
//  - pass 1: online softmax stats; pass 2: recompute, write normalized weights
//    (single HBM touch), LDS C->A transpose, context WMMA.
// ---------------------------------------------------------------------------
__global__ __launch_bounds__(256) void attn_kernel(
    const _Float16* __restrict__ Qf, const _Float16* __restrict__ Kf,
    const _Float16* __restrict__ Vt, const int* __restrict__ mask,
    float* __restrict__ AW, _Float16* __restrict__ Ctx) {
  __shared__ __align__(16) _Float16 Kls[3][32 * 72];
  __shared__ __align__(16) _Float16 Vls[3][64 * 40];
  __shared__ __align__(16) _Float16 Plds[8][16 * 40];

  const int t = threadIdx.x, lane = t & 31, wave = t >> 5;
  const int l16 = lane & 15, half = lane >> 4;
  const int bh = blockIdx.y, b = bh >> 3, h = bh & 7;
  const int qbase = blockIdx.x * 128 + wave * 16;
  const int NCH = S_ / 32;  // 64 chunks of 32 keys

  const _Float16* Qh = Qf + ((size_t)bh << 17);
  const _Float16* Kh = Kf + ((size_t)bh << 17);
  const _Float16* Vh = Vt + ((size_t)bh << 17);  // [d][s]
  const int* mrow = mask + b * S_;
  float* awh = AW + ((size_t)bh << 22);

  // Q A-fragments, resident for both passes
  const _Float16* qp = Qh + (size_t)(qbase + l16) * HD_;
  const int k0l = half * 8;
  v16h qa0 = cat8(*(const v8h*)(qp + k0l), *(const v8h*)(qp + k0l + 16));
  v16h qa1 = cat8(*(const v8h*)(qp + k0l + 32), *(const v8h*)(qp + k0l + 48));

  float mrun[8], lrun[8];
#pragma unroll
  for (int i = 0; i < 8; ++i) { mrun[i] = -3.0e38f; lrun[i] = 0.f; }

  // ---- pass 1: stats (K tiles via TDM) ----
  if (wave == 0) {
    tdm_load_2d_f16(Kh,        lds_off_of(&Kls[0][0]), 64, 32, 64, S_, 64, 4, 3);
    tdm_load_2d_f16(Kh + 2048, lds_off_of(&Kls[1][0]), 64, 32, 64, S_, 64, 4, 3);
  }
  for (int c = 0; c < NCH; ++c) {
    if (wave == 0) {
      if (c + 1 < NCH) __builtin_amdgcn_s_wait_tensorcnt(1);
      else             __builtin_amdgcn_s_wait_tensorcnt(0);
    }
    __syncthreads();  // publish K tile c; fences compute of c-1 before reuse
    if (wave == 0 && c + 2 < NCH)
      tdm_load_2d_f16(Kh + (size_t)(c + 2) * 32 * HD_,
                      lds_off_of(&Kls[(c + 2) % 3][0]), 64, 32, 64, S_, 64, 4, 3);
    const _Float16* Kb = &Kls[c % 3][0];
    int kb = c * 32;
#pragma unroll
    for (int sub = 0; sub < 2; ++sub) {
      int r = sub * 16 + l16;
      int kk0 = half * 8;
      const _Float16* kp = Kb + r * 72;
      v16h b0 = cat8(*(const v8h*)(kp + kk0), *(const v8h*)(kp + kk0 + 16));
      v16h b1 = cat8(*(const v8h*)(kp + kk0 + 32), *(const v8h*)(kp + kk0 + 48));
      v8f cc = ZERO8;
      cc = wmma_f16(qa0, b0, cc);
      cc = wmma_f16(qa1, b1, cc);
      int mv = mrow[kb + r];
#pragma unroll
      for (int i = 0; i < 8; ++i) {
        float s = mv ? cc[i] * 0.125f : -3.0e38f;
        float tm = redmax16(s);
        float mn = fmaxf(mrun[i], tm);
        float p = mv ? __expf(s - mn) : 0.f;
        float ts = redsum16(p);
        lrun[i] = lrun[i] * __expf(mrun[i] - mn) + ts;
        mrun[i] = mn;
      }
    }
  }
  float invl[8];
#pragma unroll
  for (int i = 0; i < 8; ++i) invl[i] = 1.f / lrun[i];

  __syncthreads();  // pass boundary: all waves done with pass-1 K buffers

  // ---- pass 2: normalized weights + context (K and V tiles via TDM) ----
  v8f ctx[4] = {ZERO8, ZERO8, ZERO8, ZERO8};
  _Float16* Pw = &Plds[wave][0];

  if (wave == 0) {
    tdm_load_2d_f16(Kh,        lds_off_of(&Kls[0][0]), 64, 32, 64, S_, 64, 4, 3);
    tdm_load_2d_f16(Vh,        lds_off_of(&Vls[0][0]), 32, 64, S_, 64, S_, 3, 3);
    tdm_load_2d_f16(Kh + 2048, lds_off_of(&Kls[1][0]), 64, 32, 64, S_, 64, 4, 3);
    tdm_load_2d_f16(Vh + 32,   lds_off_of(&Vls[1][0]), 32, 64, S_, 64, S_, 3, 3);
  }
  for (int c = 0; c < NCH; ++c) {
    if (wave == 0) {
      if (c + 1 < NCH) __builtin_amdgcn_s_wait_tensorcnt(2);
      else             __builtin_amdgcn_s_wait_tensorcnt(0);
    }
    __syncthreads();
    if (wave == 0 && c + 2 < NCH) {
      tdm_load_2d_f16(Kh + (size_t)(c + 2) * 32 * HD_,
                      lds_off_of(&Kls[(c + 2) % 3][0]), 64, 32, 64, S_, 64, 4, 3);
      tdm_load_2d_f16(Vh + (c + 2) * 32,
                      lds_off_of(&Vls[(c + 2) % 3][0]), 32, 64, S_, 64, S_, 3, 3);
    }
    const _Float16* Kb = &Kls[c % 3][0];
    const _Float16* Vb = &Vls[c % 3][0];
    int kb = c * 32;
#pragma unroll
    for (int sub = 0; sub < 2; ++sub) {
      int r = sub * 16 + l16;
      int kk0 = half * 8;
      const _Float16* kp = Kb + r * 72;
      v16h b0 = cat8(*(const v8h*)(kp + kk0), *(const v8h*)(kp + kk0 + 16));
      v16h b1 = cat8(*(const v8h*)(kp + kk0 + 32), *(const v8h*)(kp + kk0 + 48));
      v8f cc = ZERO8;
      cc = wmma_f16(qa0, b0, cc);
      cc = wmma_f16(qa1, b1, cc);
      int mv = mrow[kb + r];
#pragma unroll
      for (int i = 0; i < 8; ++i) {
        float p = mv ? __expf(cc[i] * 0.125f - mrun[i]) * invl[i] : 0.f;
        int row = qbase + half * 8 + i;
        awh[(size_t)row * S_ + kb + sub * 16 + l16] = p;         // final weights
        Pw[(half * 8 + i) * 40 + sub * 16 + l16] = (_Float16)p;  // C->A transpose
      }
    }
    asm volatile("s_wait_dscnt 0" ::: "memory");  // same-wave LDS RAW fence

    const _Float16* pr = Pw + l16 * 40;
    v16h pa = cat8(*(const v8h*)(pr + k0l), *(const v8h*)(pr + k0l + 16));
#pragma unroll
    for (int nt = 0; nt < 4; ++nt) {
      const _Float16* vp = Vb + (nt * 16 + l16) * 40 + half * 8;
      v16h vb = cat8(*(const v8h*)vp, *(const v8h*)(vp + 16));
      ctx[nt] = wmma_f16(pa, vb, ctx[nt]);
    }
  }

#pragma unroll
  for (int nt = 0; nt < 4; ++nt) {
#pragma unroll
    for (int i = 0; i < 8; ++i) {
      int row = qbase + half * 8 + i;
      Ctx[((size_t)b * S_ + row) * H_ + h * HD_ + nt * 16 + l16] =
          (_Float16)(ctx[nt][i]);
    }
  }
}

// ---------------------------------------------------------------------------
// Kernel 3: Ctx(f16 8192x512) @ Wo^T + bo -> f32. Ctx tiles are TDM-staged and
// double-buffered (prefetch of tile k+1 overlaps WMMA on tile k).
// ---------------------------------------------------------------------------
__global__ __launch_bounds__(256) void out_proj_kernel(
    const _Float16* __restrict__ Ctx, const float* __restrict__ Wo,
    const float* __restrict__ bo, float* __restrict__ Out) {
  __shared__ __align__(16) _Float16 Xs[2][128 * 40];
  __shared__ __align__(16) _Float16 Ws[64 * 40];

  const int t = threadIdx.x, lane = t & 31, wave = t >> 5;
  const int l16 = lane & 15, half = lane >> 4;
  const int Mbase = blockIdx.x * 128, Nbase = blockIdx.y * 64;

  v8f acc[4] = {ZERO8, ZERO8, ZERO8, ZERO8};

  if (wave == 0)
    tdm_load_2d_f16(Ctx + (size_t)Mbase * H_, lds_off_of(&Xs[0][0]),
                    32, 128, H_, MTOT_, H_, 3, 3);

  for (int kt = 0; kt < 16; ++kt) {
    int k0 = kt * 32;
#pragma unroll
    for (int rr = 0; rr < 2; ++rr) {  // stage Wo tile f32->f16
      int row = rr * 32 + (t >> 3);
      int c4 = (t & 7) * 4;
      v4f x = *(const v4f*)&Wo[(size_t)(Nbase + row) * H_ + k0 + c4];
      v4h hx;
#pragma unroll
      for (int i = 0; i < 4; ++i) hx[i] = (_Float16)x[i];
      *(v4h*)&Ws[row * 40 + c4] = hx;
    }
    if (wave == 0) __builtin_amdgcn_s_wait_tensorcnt(0);
    __syncthreads();  // publish Xs[kt&1] + Ws
    if (wave == 0 && kt + 1 < 16)
      tdm_load_2d_f16(Ctx + (size_t)Mbase * H_ + k0 + 32,
                      lds_off_of(&Xs[(kt + 1) & 1][0]), 32, 128, H_, MTOT_, H_, 3, 3);

    const _Float16* Xb = &Xs[kt & 1][0];
    int ar = wave * 16 + l16;
    int akl = half * 8;
    v16h a = cat8(*(const v8h*)&Xb[ar * 40 + akl],
                  *(const v8h*)&Xb[ar * 40 + akl + 16]);
#pragma unroll
    for (int nt = 0; nt < 4; ++nt) {
      int br = nt * 16 + l16;
      int bkl = half * 8;
      v16h bfr = cat8(*(const v8h*)&Ws[br * 40 + bkl],
                      *(const v8h*)&Ws[br * 40 + bkl + 16]);
      acc[nt] = wmma_f16(a, bfr, acc[nt]);
    }
    __syncthreads();  // protect Ws (and Xs buffer rotation) before next stage
  }

#pragma unroll
  for (int nt = 0; nt < 4; ++nt) {
    int n = Nbase + nt * 16 + l16;
    float bv = bo[n];
#pragma unroll
    for (int i = 0; i < 8; ++i) {
      int M = Mbase + wave * 16 + half * 8 + i;
      Out[(size_t)M * H_ + n] = acc[nt][i] + bv;
    }
  }
}

// ---------------------------------------------------------------------------
extern "C" void kernel_launch(void* const* d_in, const int* in_sizes, int n_in,
                              void* d_out, int out_size, void* d_ws,
                              size_t ws_size, hipStream_t stream) {
  (void)in_sizes; (void)n_in; (void)out_size; (void)ws_size;
  const float* values = (const float*)d_in[0];
  const float* keys   = (const float*)d_in[1];
  const float* query  = (const float*)d_in[2];
  const int*   mask   = (const int*)d_in[3];
  const float* Wq = (const float*)d_in[4];  const float* bq = (const float*)d_in[5];
  const float* Wk = (const float*)d_in[6];  const float* bk = (const float*)d_in[7];
  const float* Wv = (const float*)d_in[8];  const float* bv = (const float*)d_in[9];
  const float* Wo = (const float*)d_in[10]; const float* bo = (const float*)d_in[11];

  float* out = (float*)d_out;                  // (4,2048,512)
  float* aw  = out + (size_t)B_ * S_ * H_;     // (4,8,2048,2048)

  char* ws = (char*)d_ws;
  _Float16* Qf  = (_Float16*)(ws);
  _Float16* Kf  = (_Float16*)(ws + ((size_t)8 << 20));
  _Float16* Vt  = (_Float16*)(ws + ((size_t)16 << 20));
  _Float16* Ctx = (_Float16*)(ws + ((size_t)24 << 20));

  dim3 blk(256);
  dim3 gProj(MTOT_ / 128, H_ / 64);   // 64 x 8
  qkv_proj_kernel<<<gProj, blk, 0, stream>>>(query,  Wq, bq, Qf, 0);
  qkv_proj_kernel<<<gProj, blk, 0, stream>>>(keys,   Wk, bk, Kf, 0);
  qkv_proj_kernel<<<gProj, blk, 0, stream>>>(values, Wv, bv, Vt, 1);

  dim3 gAttn(S_ / 128, B_ * NH_);     // 16 x 32
  attn_kernel<<<gAttn, blk, 0, stream>>>(Qf, Kf, Vt, mask, aw, Ctx);

  out_proj_kernel<<<gProj, blk, 0, stream>>>(Ctx, Wo, bo, out);
}
